// TransformerBlock_13846974562947
// MI455X (gfx1250) — matrix operand order repaired and checked by reading, not compiled, not run on hardware
//
#include <hip/hip_runtime.h>
#include <math.h>

// ---------------------------------------------------------------------------
// Types for CDNA5 WMMA (gfx1250, wave32)
// ---------------------------------------------------------------------------
typedef __attribute__((ext_vector_type(16))) __bf16 v16bf;
typedef __attribute__((ext_vector_type(8)))  __bf16 v8bf;
typedef __attribute__((ext_vector_type(8)))  float  v8f;
typedef __attribute__((ext_vector_type(4)))  float  v4f;
typedef __attribute__((ext_vector_type(4)))  unsigned int v4u;
typedef __attribute__((ext_vector_type(8)))  int    v8i;
typedef __attribute__((ext_vector_type(4)))  int    v4i;
typedef __bf16 bf16_t;

#define DMODEL 1024
#define NHEAD  16
#define HDIM   64
#define SEQ    2048
#define BATCH  2
#define FFDIM  4096
#define NTOK   (BATCH*SEQ)   // 4096 rows

#if defined(__has_builtin)
#if __has_builtin(__builtin_amdgcn_tensor_load_to_lds) && \
    __has_builtin(__builtin_amdgcn_s_wait_tensorcnt)
#define HAVE_TDM 1
#endif
#endif
#ifndef HAVE_TDM
#define HAVE_TDM 0
#endif

// ---------------------------------------------------------------------------
// Fragment loader, contiguous k order (B fragments use the same order):
//   A (16x32 bf16): lane l: m=l&15, half=l>>4; elem e -> K = half*16+e
//                   => two contiguous 16B loads
// ---------------------------------------------------------------------------
__device__ __forceinline__ v16bf load_a_frag(const bf16_t* __restrict__ base,
                                             int lda, int m, int k0, int half) {
  const bf16_t* p = base + (size_t)m * lda + k0 + half * 16;
  v8bf lo = *(const v8bf*)(p);
  v8bf hi = *(const v8bf*)(p + 8);
  v16bf a;
#pragma unroll
  for (int i = 0; i < 8; ++i) { a[i] = lo[i]; a[i + 8] = hi[i]; }
  return a;
}

// ---------------------------------------------------------------------------
// TDM: DMA a 64(rows) x 64(K) bf16 tile of the [N,K] weight matrix into LDS.
// Descriptor bit packing per CDNA5 ISA:
//   group0: w0=count(1) | is_restore(0) | gather(0); w1=lds_addr;
//           w2=global_addr[31:0]; w3=global_addr[56:32] | type(2)<<30
//   group1: w0=workgroup_mask(0) | data_size(1=2B)<<16; tensor_dim0/1 = 64;
//           tile_dim0/1 = 64, tile_dim2 = 0; tensor_dim0_stride = K elems
// Tracked with TENSORcnt; issued by one wave, published via barrier.
// ---------------------------------------------------------------------------
#if HAVE_TDM
__device__ __forceinline__ void tdm_load_tile(unsigned lds_off,
                                              const bf16_t* gptr,
                                              unsigned strideK) {
  unsigned long long ga = (unsigned long long)(uintptr_t)gptr;
  v4u g0;
  g0[0] = 1u;                                   // count=1 (valid user D#)
  g0[1] = lds_off;                              // lds_addr (bytes)
  g0[2] = (unsigned)ga;                         // global_addr[31:0]
  g0[3] = (unsigned)(ga >> 32) | (2u << 30);    // global_addr[56:32] | type=2
  v8i g1;
  g1[0] = (int)(1u << 16);                      // data_size=1 -> 2 bytes
  g1[1] = (int)(64u << 16);                     // tensor_dim0[15:0]=64
  g1[2] = (int)(64u << 16);                     // dim0[31:16]=0, tensor_dim1=64
  g1[3] = (int)(64u << 16);                     // dim1[31:16]=0, tile_dim0=64
  g1[4] = 64;                                   // tile_dim1=64, tile_dim2=0
  g1[5] = (int)strideK;                         // tensor_dim0_stride[31:0]
  g1[6] = 0;                                    // stride[47:32]=0, dim1_stride=0
  g1[7] = 0;
  v4i z = {0, 0, 0, 0};
#if __clang_major__ >= 23
  v8i z8 = {0, 0, 0, 0, 0, 0, 0, 0};
  __builtin_amdgcn_tensor_load_to_lds(g0, g1, z, z, z8, 0);
#else
  __builtin_amdgcn_tensor_load_to_lds(g0, g1, z, z, 0);
#endif
}
#endif

// ---------------------------------------------------------------------------
// GEMM: C[m,n] = sum_k A[m,k] * W[n,k] + bias[n]  (+relu) (+residual)
// A: [M,K] bf16 row-major; W: [N,K] bf16 row-major.
// Block = 256 threads = 8 waves; block tile 128(M) x 64(N); K-step 64.
// The 64x64 weight tile (shared by all 8 waves) is staged into LDS by the
// Tensor Data Mover, double-buffered: TDM fills buf^1 while waves run 8
// WMMAs against buf; s_wait_tensorcnt + barrier republish per iteration.
// All 8 B fragments + 2 A fragments are preloaded per tile so the 8 WMMAs
// issue back-to-back behind a single wait instead of load/wait/wmma x8.
// ---------------------------------------------------------------------------
__global__ __launch_bounds__(256)
void gemm_bf16_wmma(const bf16_t* __restrict__ A, const bf16_t* __restrict__ W,
                    const float* __restrict__ bias,
                    const float* __restrict__ residual,
                    void* __restrict__ Cout, int M, int N, int K, int flags) {
  __shared__ __align__(16) bf16_t Bt[2][64 * 64];   // 2 x 8KB
  const int lane = threadIdx.x & 31;
  const int wave = threadIdx.x >> 5;
  const int m0   = blockIdx.y * 128 + wave * 16;
  const int n0   = blockIdx.x * 64;
  const int nc   = lane & 15;
  const int half = lane >> 4;
  const int klo  = half * 16;
  const int nk   = K >> 6;

  const bf16_t* wbase = W + (size_t)n0 * K;

  // ---- stage tile 0 ----
#if HAVE_TDM
  if (threadIdx.x < 32) {
    tdm_load_tile((unsigned)(uintptr_t)&Bt[0][0], wbase, (unsigned)K);
    __builtin_amdgcn_s_wait_tensorcnt(0);
  }
#else
  {
    const int row = threadIdx.x >> 2;          // 0..63
    const int col = (threadIdx.x & 3) * 16;    // 0,16,32,48
    *(v16bf*)(&Bt[0][row * 64 + col]) =
        *(const v16bf*)(wbase + (size_t)row * K + col);
  }
#endif
  __syncthreads();

  v8f acc[4] = {};

  for (int kt = 0; kt < nk; ++kt) {
    const int cur = kt & 1;
    // ---- kick off DMA of next K-tile into the other buffer ----
    if (kt + 1 < nk) {
#if HAVE_TDM
      if (threadIdx.x < 32)
        tdm_load_tile((unsigned)(uintptr_t)&Bt[cur ^ 1][0],
                      wbase + (kt + 1) * 64, (unsigned)K);
#else
      const int row = threadIdx.x >> 2;
      const int col = (threadIdx.x & 3) * 16;
      *(v16bf*)(&Bt[cur ^ 1][row * 64 + col]) =
          *(const v16bf*)(wbase + (size_t)row * K + (kt + 1) * 64 + col);
#endif
      __builtin_prefetch(A + (size_t)(m0 + nc) * K + (kt + 1) * 64, 0, 1);
    }
    // ---- preload both A fragments and all 8 B fragments ----
    v16bf a0 = load_a_frag(A, K, m0 + nc, kt * 64, half);
    v16bf a1 = load_a_frag(A, K, m0 + nc, kt * 64 + 32, half);
    v16bf bf[8];
#pragma unroll
    for (int c = 0; c < 2; ++c)
#pragma unroll
      for (int j = 0; j < 4; ++j)
        bf[c * 4 + j] =
            *(const v16bf*)(&Bt[cur][(j * 16 + nc) * 64 + c * 32 + klo]);
    // ---- 8 back-to-back WMMAs ----
#pragma unroll
    for (int j = 0; j < 4; ++j)
      acc[j] = __builtin_amdgcn_wmma_f32_16x16x32_bf16(
          false, a0, false, bf[j], (short)0, acc[j], false, false);
#pragma unroll
    for (int j = 0; j < 4; ++j)
      acc[j] = __builtin_amdgcn_wmma_f32_16x16x32_bf16(
          false, a1, false, bf[4 + j], (short)0, acc[j], false, false);
#if HAVE_TDM
    if (threadIdx.x < 32) __builtin_amdgcn_s_wait_tensorcnt(0);
#endif
    __syncthreads();
  }

  const bool out_bf = flags & 1;
  const bool relu   = flags & 2;
#pragma unroll
  for (int j = 0; j < 4; ++j) {
    const int   col = n0 + j * 16 + nc;
    const float bv  = bias ? bias[col] : 0.f;
#pragma unroll
    for (int r = 0; r < 8; ++r) {
      const int m = m0 + r + 8 * half;
      float c = acc[j][r] + bv;
      if (relu) c = fmaxf(c, 0.f);
      if (residual) c += residual[(size_t)m * N + col];
      if (out_bf) ((bf16_t*)Cout)[(size_t)m * N + col] = (bf16_t)c;
      else        ((float*)Cout)[(size_t)m * N + col]  = c;
    }
  }
}

// ---------------------------------------------------------------------------
// LayerNorm (D=1024) -> bf16. One row per 256-thread block, 4 elems/thread.
// ---------------------------------------------------------------------------
__global__ __launch_bounds__(256)
void layernorm_to_bf16(const float* __restrict__ x, const float* __restrict__ g,
                       const float* __restrict__ beta, bf16_t* __restrict__ out) {
  __shared__ float red[2][8];
  const int row = blockIdx.x;
  const float* xr = x + (size_t)row * DMODEL;
  v4f v = *(const v4f*)(xr + threadIdx.x * 4);
  float s  = v.x + v.y + v.z + v.w;
  float sq = v.x * v.x + v.y * v.y + v.z * v.z + v.w * v.w;
#pragma unroll
  for (int off = 16; off; off >>= 1) {
    s  += __shfl_xor(s,  off);
    sq += __shfl_xor(sq, off);
  }
  if ((threadIdx.x & 31) == 0) {
    red[0][threadIdx.x >> 5] = s;
    red[1][threadIdx.x >> 5] = sq;
  }
  __syncthreads();
  float ts = 0.f, tq = 0.f;
#pragma unroll
  for (int i = 0; i < 8; ++i) { ts += red[0][i]; tq += red[1][i]; }
  const float mean = ts * (1.f / DMODEL);
  const float var  = tq * (1.f / DMODEL) - mean * mean;
  const float rstd = rsqrtf(var + 1e-5f);
#pragma unroll
  for (int i = 0; i < 4; ++i) {
    const int c = threadIdx.x * 4 + i;
    out[(size_t)row * DMODEL + c] = (bf16_t)((v[i] - mean) * rstd * g[c] + beta[c]);
  }
}

// ---------------------------------------------------------------------------
// f32 -> bf16 elementwise
// ---------------------------------------------------------------------------
__global__ void f32_to_bf16(const float* __restrict__ in, bf16_t* __restrict__ out,
                            int n) {
  const int i = blockIdx.x * blockDim.x + threadIdx.x;
  if (i < n) out[i] = (bf16_t)in[i];
}

// ---------------------------------------------------------------------------
// Flash attention: one wave owns 16 query rows of one (batch, head).
// Online softmax over key blocks of 32. Per block:
//   preload 4 K frags -> 4 WMMAs (scores), P restaged via LDS (C/D -> A
//   layout) with explicit s_wait_dscnt, preload 4 V frags -> 4 WMMAs (P@V).
// qkv layout per token row (3072): [h*192 + {q:0..63, k:64..127, v:128..191}]
// ---------------------------------------------------------------------------
__global__ __launch_bounds__(256)
void attn_flash_wmma(const bf16_t* __restrict__ qkv, bf16_t* __restrict__ ctxb) {
  __shared__ __align__(16) bf16_t Pbuf[8][16 * 32];  // 1KB per wave
  const int lane = threadIdx.x & 31;
  const int wave = threadIdx.x >> 5;
  const int gw   = blockIdx.x * 8 + wave;     // 4096 waves total
  const int qt   = gw & 127;                  // S/16 = 128 query tiles
  const int h    = (gw >> 7) & 15;
  const int b    = gw >> 11;
  const int q0   = qt * 16;
  const int nc   = lane & 15;
  const int half = lane >> 4;
  const int klo  = half * 16;
  const float scale = 0.125f;                 // 1/sqrt(64)

  const int rowstride = 3 * DMODEL;           // 3072
  const bf16_t* base = qkv + (size_t)(b * SEQ) * rowstride + h * (3 * HDIM);
  bf16_t* pw = &Pbuf[wave][0];

  v16bf qa[2];
#pragma unroll
  for (int c = 0; c < 2; ++c)
    qa[c] = load_a_frag(base, rowstride, q0 + nc, c * 32, half);

  float mrow[8], lrow[8];
  v8f acc[4] = {};
#pragma unroll
  for (int r = 0; r < 8; ++r) { mrow[r] = -INFINITY; lrow[r] = 0.f; }

  for (int kb = 0; kb < SEQ; kb += 32) {
    // ---- preload all 4 K fragments, then 4 back-to-back WMMAs ----
    v16bf kfr[4];
#pragma unroll
    for (int nt = 0; nt < 2; ++nt)
#pragma unroll
      for (int c = 0; c < 2; ++c)
        kfr[nt * 2 + c] = *(const v16bf*)(base +
            (size_t)(kb + nt * 16 + nc) * rowstride + HDIM + c * 32 + klo);
    v8f sfr[2];
#pragma unroll
    for (int nt = 0; nt < 2; ++nt) {
      v8f s = {};
#pragma unroll
      for (int c = 0; c < 2; ++c)
        s = __builtin_amdgcn_wmma_f32_16x16x32_bf16(
            false, qa[c], false, kfr[nt * 2 + c], (short)0, s, false, false);
      sfr[nt] = s;
    }

#pragma unroll
    for (int r = 0; r < 8; ++r) {
      const float v0 = sfr[0][r] * scale;
      const float v1 = sfr[1][r] * scale;
      float mx = fmaxf(v0, v1);
      mx = fmaxf(mx, __shfl_xor(mx, 1));
      mx = fmaxf(mx, __shfl_xor(mx, 2));
      mx = fmaxf(mx, __shfl_xor(mx, 4));
      mx = fmaxf(mx, __shfl_xor(mx, 8));
      const float nm = fmaxf(mrow[r], mx);
      const float p0 = __expf(v0 - nm);
      const float p1 = __expf(v1 - nm);
      float rs = p0 + p1;
      rs += __shfl_xor(rs, 1);
      rs += __shfl_xor(rs, 2);
      rs += __shfl_xor(rs, 4);
      rs += __shfl_xor(rs, 8);
      const float corr = __expf(mrow[r] - nm);
      lrow[r] = lrow[r] * corr + rs;
      mrow[r] = nm;
#pragma unroll
      for (int j = 0; j < 4; ++j) acc[j][r] *= corr;
      const int m = r + 8 * half;
      pw[m * 32 + nc]      = (bf16_t)p0;
      pw[m * 32 + 16 + nc] = (bf16_t)p1;
    }
    asm volatile("s_wait_dscnt 0x0" ::: "memory");

    v16bf pa;
    {
      const bf16_t* p = pw + nc * 32 + half * 16;
      v8bf lo = *(const v8bf*)(p);
      v8bf hi = *(const v8bf*)(p + 8);
#pragma unroll
      for (int i = 0; i < 8; ++i) { pa[i] = lo[i]; pa[i + 8] = hi[i]; }
    }

    // ---- preload all 4 V fragments, then 4 back-to-back WMMAs ----
    v16bf vfr[4];
#pragma unroll
    for (int j = 0; j < 4; ++j) {
      const bf16_t* vp = base + (size_t)(kb + klo) * rowstride +
                         2 * HDIM + j * 16 + nc;
#pragma unroll
      for (int e = 0; e < 16; ++e) vfr[j][e] = vp[(size_t)e * rowstride];
    }
#pragma unroll
    for (int j = 0; j < 4; ++j)
      acc[j] = __builtin_amdgcn_wmma_f32_16x16x32_bf16(
          false, pa, false, vfr[j], (short)0, acc[j], false, false);
  }

#pragma unroll
  for (int r = 0; r < 8; ++r) {
    const float inv = 1.f / lrow[r];
    const int m = q0 + r + 8 * half;
    const size_t rowoff = (size_t)(b * SEQ + m) * DMODEL + h * HDIM;
#pragma unroll
    for (int j = 0; j < 4; ++j)
      ctxb[rowoff + j * 16 + nc] = (bf16_t)(acc[j][r] * inv);
  }
}

// ---------------------------------------------------------------------------
// Host orchestration
// ---------------------------------------------------------------------------
extern "C" void kernel_launch(void* const* d_in, const int* in_sizes, int n_in,
                              void* d_out, int out_size, void* d_ws, size_t ws_size,
                              hipStream_t stream) {
  (void)in_sizes; (void)n_in; (void)out_size; (void)ws_size;
  const float* x     = (const float*)d_in[0];
  const float* w_qkv = (const float*)d_in[1];
  const float* b_qkv = (const float*)d_in[2];
  const float* w_out = (const float*)d_in[3];
  const float* b_out = (const float*)d_in[4];
  const float* w1    = (const float*)d_in[5];
  const float* b1    = (const float*)d_in[6];
  const float* w2    = (const float*)d_in[7];
  const float* b2    = (const float*)d_in[8];
  const float* ln1g  = (const float*)d_in[9];
  const float* ln1b  = (const float*)d_in[10];
  const float* ln2g  = (const float*)d_in[11];
  const float* ln2b  = (const float*)d_in[12];
  float* out = (float*)d_out;

  // Workspace layout (80 MiB total)
  char* ws = (char*)d_ws;
  bf16_t* wqkv_bf = (bf16_t*)(ws);              // 6 MiB  (3072x1024)
  bf16_t* wout_bf = (bf16_t*)(ws + 6291456);    // 2 MiB  (1024x1024)
  bf16_t* w1_bf   = (bf16_t*)(ws + 8388608);    // 8 MiB  (4096x1024)
  bf16_t* w2_bf   = (bf16_t*)(ws + 16777216);   // 8 MiB  (1024x4096)
  bf16_t* h_bf    = (bf16_t*)(ws + 25165824);   // 8 MiB  ln1/ln2 out
  bf16_t* qkv_bf  = (bf16_t*)(ws + 33554432);   // 24 MiB (4096x3072)
  bf16_t* ctx_bf  = (bf16_t*)(ws + 58720256);   // 8 MiB  (4096x1024)
  float*  x1      = (float*) (ws + 67108864);   // 16 MiB residual after attn
  bf16_t* ffn1_bf = qkv_bf;                     // 32 MiB overlay (qkv+ctx dead)

  // 1. weights -> bf16
  f32_to_bf16<<<(3145728 + 255) / 256, 256, 0, stream>>>(w_qkv, wqkv_bf, 3145728);
  f32_to_bf16<<<(1048576 + 255) / 256, 256, 0, stream>>>(w_out, wout_bf, 1048576);
  f32_to_bf16<<<(4194304 + 255) / 256, 256, 0, stream>>>(w1,    w1_bf,   4194304);
  f32_to_bf16<<<(4194304 + 255) / 256, 256, 0, stream>>>(w2,    w2_bf,   4194304);

  // 2. LN1
  layernorm_to_bf16<<<NTOK, 256, 0, stream>>>(x, ln1g, ln1b, h_bf);

  // 3. QKV projection: [4096,1024] @ [3072,1024]^T -> bf16
  gemm_bf16_wmma<<<dim3(3072 / 64, NTOK / 128), 256, 0, stream>>>(
      h_bf, wqkv_bf, b_qkv, nullptr, qkv_bf, NTOK, 3072, DMODEL, /*bf16 out*/1);

  // 4. attention (flash, per-wave 16 query rows): 4096 waves / 8 = 512 blocks
  attn_flash_wmma<<<512, 256, 0, stream>>>(qkv_bf, ctx_bf);

  // 5. output projection + residual -> x1 (f32)
  gemm_bf16_wmma<<<dim3(DMODEL / 64, NTOK / 128), 256, 0, stream>>>(
      ctx_bf, wout_bf, b_out, x, x1, NTOK, DMODEL, DMODEL, 0);

  // 6. LN2
  layernorm_to_bf16<<<NTOK, 256, 0, stream>>>(x1, ln2g, ln2b, h_bf);

  // 7. FFN up + ReLU -> bf16
  gemm_bf16_wmma<<<dim3(FFDIM / 64, NTOK / 128), 256, 0, stream>>>(
      h_bf, w1_bf, b1, nullptr, ffn1_bf, NTOK, FFDIM, DMODEL, 1 | 2);

  // 8. FFN down + residual -> out (f32)
  gemm_bf16_wmma<<<dim3(DMODEL / 64, NTOK / 128), 256, 0, stream>>>(
      ffn1_bf, w2_bf, b2, x1, out, NTOK, DMODEL, FFDIM, 0);
}
